// DustbinRetrievaler_49993419325681
// MI455X (gfx1250) — compile-verified, matrix operand order checked by reading, not run on hardware
//
#include <hip/hip_runtime.h>
#include <hip/hip_bf16.h>
#include <math.h>

// ---- problem constants (from reference) ----
#define B_   64
#define K_   1024
#define C_   512
#define D_   256
#define L_   2
#define H_   4
#define FF_  1024
#define DH_  64
#define S_   1026            // 2 global tokens + K candidates
#define MS_  (B_*S_)         // 65664 rows (divisible by 128)
#define SCALE_ 0.125f        // 1/sqrt(DH)

typedef __attribute__((ext_vector_type(8)))  float          v8f;
typedef __attribute__((ext_vector_type(16))) __bf16         v16bf;
typedef __attribute__((ext_vector_type(8)))  unsigned short u16x8;
typedef __attribute__((ext_vector_type(16))) unsigned short u16x16;

__device__ __forceinline__ unsigned short f2bf(float f) {
  unsigned u = __builtin_bit_cast(unsigned, f);
  u += 0x7FFFu + ((u >> 16) & 1u);          // round-to-nearest-even
  return (unsigned short)(u >> 16);
}
__device__ __forceinline__ float bf2f(unsigned short h) {
  return __builtin_bit_cast(float, (unsigned)h << 16);
}
__device__ __forceinline__ u16x8 pk_bf(float4 a, float4 b) {
  u16x8 r;
  r[0]=f2bf(a.x); r[1]=f2bf(a.y); r[2]=f2bf(a.z); r[3]=f2bf(a.w);
  r[4]=f2bf(b.x); r[5]=f2bf(b.y); r[6]=f2bf(b.z); r[7]=f2bf(b.w);
  return r;
}

// Load one 16x32 bf16 fragment for this lane from an LDS tile row.
// Lane layout per ISA: lane%16 = row, kb = (lane/16)*8; VGPR0..3 <- K[kb..kb+7],
// VGPR4..7 <- K[kb+16..kb+23].  Row stride 72 u16 (144 B, 16B-aligned).
__device__ __forceinline__ v16bf ldfrag(const unsigned short* rowp, int kb) {
  u16x8 lo = *(const u16x8*)(rowp + kb);
  u16x8 hi = *(const u16x8*)(rowp + kb + 16);
  u16x16 c = __builtin_shufflevector(lo, hi, 0,1,2,3,4,5,6,7,8,9,10,11,12,13,14,15);
  return __builtin_bit_cast(v16bf, c);
}

#define ASTRIDE 72
#define BOFF    (128 * ASTRIDE)          // B tile starts after 128-row A tile
#define ABSZ    (BOFF + 64 * ASTRIDE)    // 13824 u16 per buffer (27648 B)

// -------------------------------------------------------------------------
// Tiled WMMA GEMM:  Y[orow, n] = sum_k A[row,k] * W[n,k]  (+ epilogue)
//   EPI: 0 = +bias ; 1 = +bias+bias2 ; 2 = gelu(acc+bias) ; 3 = +bias+resid
//   Block tile 128x64, BK=64, 8 waves, wave tile 32x32 (2x2 WMMA 16x16x32).
//   Double-buffered LDS: one barrier per k-chunk; next chunk's global loads
//   are issued before the current chunk's WMMA burst.
//   REMAP (compile-time): orow = (row>>10)*1026 + 2 + (row&1023)  (embed scatter)
// -------------------------------------------------------------------------
template<bool ABF16, int EPI, bool OBF16, bool REMAP>
__global__ __launch_bounds__(256)
void gemm_wmma(const void* __restrict__ Av, const float* __restrict__ W,
               const float* __restrict__ bias, const float* __restrict__ bias2,
               const float* __restrict__ resid, void* __restrict__ Ov,
               int N, int Kd)
{
  __shared__ __align__(16) unsigned short lds[2][ABSZ];   // 54 KB of 320 KB/WGP

  const int tid  = threadIdx.x;
  const int lane = tid & 31;
  const int wave = tid >> 5;
  const int wm   = wave & 3;          // wave m-tile (0..3) -> rows wm*32
  const int wn   = wave >> 2;         // wave n-tile (0..1) -> cols wn*32
  const int mBase = blockIdx.y * 128;
  const int nBase = blockIdx.x * 64;

  v8f acc[2][2] = {};

  const int arow = tid >> 1, ahalf = tid & 1;   // A tile: 128 rows x (2 x 32k)
  const int brow = tid >> 2, bq    = tid & 3;   // B tile:  64 rows x (4 x 16k)
  const size_t aGRow = (size_t)(mBase + arow);
  const size_t bGRow = (size_t)(nBase + brow);

  const int fr = lane & 15;
  const int kb = (lane >> 4) * 8;
  const int aoff0 = (wm * 32      + fr) * ASTRIDE;
  const int aoff1 = (wm * 32 + 16 + fr) * ASTRIDE;
  const int boff0 = BOFF + (wn * 32      + fr) * ASTRIDE;
  const int boff1 = BOFF + (wn * 32 + 16 + fr) * ASTRIDE;
  const int asto  = arow * ASTRIDE + ahalf * 32;
  const int bsto  = BOFF + brow * ASTRIDE + bq * 16;

  // ---- register staging of one 64-k chunk ----
  auto stage = [&](int k0, u16x8 aV[4], u16x8& bV0, u16x8& bV1) {
    if (ABF16) {
      const unsigned short* ap = (const unsigned short*)Av + aGRow * Kd + k0 + ahalf * 32;
      aV[0] = ((const u16x8*)ap)[0];
      aV[1] = ((const u16x8*)ap)[1];
      aV[2] = ((const u16x8*)ap)[2];
      aV[3] = ((const u16x8*)ap)[3];
    } else {
      const float4* af = (const float4*)((const float*)Av + aGRow * Kd + k0 + ahalf * 32);
      aV[0] = pk_bf(af[0], af[1]);
      aV[1] = pk_bf(af[2], af[3]);
      aV[2] = pk_bf(af[4], af[5]);
      aV[3] = pk_bf(af[6], af[7]);
    }
    const float4* bf = (const float4*)(W + bGRow * Kd + k0 + bq * 16);
    bV0 = pk_bf(bf[0], bf[1]);
    bV1 = pk_bf(bf[2], bf[3]);
  };
  auto commit = [&](unsigned short* buf, const u16x8 aV[4], u16x8 bV0, u16x8 bV1) {
    unsigned short* as = buf + asto;
    ((u16x8*)as)[0] = aV[0]; ((u16x8*)as)[1] = aV[1];
    ((u16x8*)as)[2] = aV[2]; ((u16x8*)as)[3] = aV[3];
    unsigned short* bs = buf + bsto;
    ((u16x8*)bs)[0] = bV0;   ((u16x8*)bs)[1] = bV1;
  };

  const int nch = Kd >> 6;            // Kd in {256,512,1024} -> 4/8/16 chunks

  // prologue: chunk 0 into buffer 0
  {
    u16x8 aV[4], bV0, bV1;
    stage(0, aV, bV0, bV1);
    commit(&lds[0][0], aV, bV0, bV1);
  }
  __syncthreads();

  for (int i = 0; i < nch; ++i) {
    const unsigned short* buf = &lds[i & 1][0];
    const bool more = (i + 1) < nch;

    // issue next chunk's global loads before this chunk's WMMAs
    u16x8 aV[4], bV0, bV1;
    if (more) stage((i + 1) << 6, aV, bV0, bV1);

    #pragma unroll
    for (int sub = 0; sub < 2; ++sub) {
      const int o = sub * 32;
      v16bf a0 = ldfrag(buf + aoff0 + o, kb);
      v16bf a1 = ldfrag(buf + aoff1 + o, kb);
      v16bf b0 = ldfrag(buf + boff0 + o, kb);
      v16bf b1 = ldfrag(buf + boff1 + o, kb);
      acc[0][0] = __builtin_amdgcn_wmma_f32_16x16x32_bf16(false, a0, false, b0, (short)0, acc[0][0], false, false);
      acc[0][1] = __builtin_amdgcn_wmma_f32_16x16x32_bf16(false, a0, false, b1, (short)0, acc[0][1], false, false);
      acc[1][0] = __builtin_amdgcn_wmma_f32_16x16x32_bf16(false, a1, false, b0, (short)0, acc[1][0], false, false);
      acc[1][1] = __builtin_amdgcn_wmma_f32_16x16x32_bf16(false, a1, false, b1, (short)0, acc[1][1], false, false);
    }

    if (more) {
      // safe: all reads of lds[(i+1)&1] finished before the barrier that ended
      // iteration i-1 (syncthreads waits each wave's DScnt before signaling)
      commit(&lds[(i + 1) & 1][0], aV, bV0, bV1);
      __syncthreads();
    }
  }

  // ---- epilogue:  C/D layout: VGPR g -> row g + (lane/16)*8, col lane%16 ----
  const int col16 = lane & 15;
  const int rsel  = (lane >> 4) * 8;
  #pragma unroll
  for (int nt = 0; nt < 2; ++nt) {
    const int gcol = nBase + wn * 32 + nt * 16 + col16;
    const float bv  = bias[gcol];
    const float b2v = (EPI == 1) ? bias2[gcol] : 0.f;
    #pragma unroll
    for (int mt = 0; mt < 2; ++mt) {
      #pragma unroll
      for (int g = 0; g < 8; ++g) {
        const int grow = mBase + wm * 32 + mt * 16 + rsel + g;
        float vv = acc[mt][nt][g] + bv;
        if (EPI == 1) vv += b2v;
        if (EPI == 2) vv = 0.5f * vv * (1.0f + erff(vv * 0.70710678118654752f));
        const int orow = REMAP ? ((grow >> 10) * S_ + 2 + (grow & 1023)) : grow;
        const size_t oidx = (size_t)orow * N + gcol;
        if (EPI == 3) vv += resid[oidx];
        if (OBF16) ((unsigned short*)Ov)[oidx] = f2bf(vv);
        else       ((float*)Ov)[oidx] = vv;
      }
    }
  }
}

// ---- query + dustbin token embed (tiny) ----
__global__ __launch_bounds__(256)
void embed_small(const float* __restrict__ query, const float* __restrict__ qp_w,
                 const float* __restrict__ qp_b, const float* __restrict__ dustbin,
                 const float* __restrict__ te, float* __restrict__ x)
{
  int b = blockIdx.x, d = threadIdx.x;       // blockDim = 256 = D
  float acc = qp_b[d] + te[d];
  const float* qr = query + (size_t)b * C_;
  const float* wr = qp_w  + (size_t)d * C_;
  for (int c = 0; c < C_; ++c) acc += qr[c] * wr[c];
  x[(size_t)(b * S_    ) * D_ + d] = acc;
  x[(size_t)(b * S_ + 1) * D_ + d] = dustbin[d] + te[D_ + d];
}

// ---- wave-per-row LayerNorm (D=256), fp32 in -> bf16 out ----
__global__ __launch_bounds__(256)
void ln_rows(const float* __restrict__ x, const float* __restrict__ w,
             const float* __restrict__ b, unsigned short* __restrict__ out, int rows)
{
  int wid  = (blockIdx.x * 256 + threadIdx.x) >> 5;
  int lane = threadIdx.x & 31;
  if (wid >= rows) return;
  const float* xr = x + (size_t)wid * D_;
  float vals[8], sm = 0.f, sq = 0.f;
  #pragma unroll
  for (int i = 0; i < 8; ++i) { float t = xr[lane + 32 * i]; vals[i] = t; sm += t; sq += t * t; }
  #pragma unroll
  for (int off = 16; off >= 1; off >>= 1) { sm += __shfl_xor(sm, off, 32); sq += __shfl_xor(sq, off, 32); }
  float mean = sm * (1.f / D_);
  float var  = sq * (1.f / D_) - mean * mean;
  float inv  = rsqrtf(var + 1e-5f);
  #pragma unroll
  for (int i = 0; i < 8; ++i) {
    int d = lane + 32 * i;
    out[(size_t)wid * D_ + d] = f2bf((vals[i] - mean) * inv * w[d] + b[d]);
  }
}

// ---- global tokens (t=0,1) attend to all S keys; one block per (t,h,b) ----
__global__ __launch_bounds__(256)
void attn_global(const unsigned short* __restrict__ q, const unsigned short* __restrict__ k,
                 const unsigned short* __restrict__ v, const int* __restrict__ cmask,
                 unsigned short* __restrict__ attn)
{
  const int t = blockIdx.x, h = blockIdx.y, b = blockIdx.z;
  const int tid = threadIdx.x;
  __shared__ float sc[S_];
  __shared__ float qsh[DH_];
  __shared__ float red[256];

  if (tid < DH_) qsh[tid] = bf2f(q[(size_t)(b * S_ + t) * D_ + h * DH_ + tid]);
  __syncthreads();

  float lmax = -1e30f;
  for (int s = tid; s < S_; s += 256) {
    const unsigned short* kr = k + (size_t)(b * S_ + s) * D_ + h * DH_;
    float d = 0.f;
    #pragma unroll 8
    for (int i = 0; i < DH_; ++i) d += qsh[i] * bf2f(kr[i]);
    d *= SCALE_;
    if (s >= 2 && cmask[b * K_ + (s - 2)] == 0) d = -1e30f;
    sc[s] = d;
    lmax = fmaxf(lmax, d);
  }
  red[tid] = lmax; __syncthreads();
  for (int st = 128; st > 0; st >>= 1) { if (tid < st) red[tid] = fmaxf(red[tid], red[tid + st]); __syncthreads(); }
  float mx = red[0]; __syncthreads();

  float lsum = 0.f;
  for (int s = tid; s < S_; s += 256) { float e = __expf(sc[s] - mx); sc[s] = e; lsum += e; }
  red[tid] = lsum; __syncthreads();
  for (int st = 128; st > 0; st >>= 1) { if (tid < st) red[tid] += red[tid + st]; __syncthreads(); }
  float inv = 1.f / red[0]; __syncthreads();

  const int d = tid & (DH_ - 1);
  const int chunk = tid >> 6;                 // 4 S-chunks
  float acc = 0.f;
  for (int s = chunk; s < S_; s += 4)
    acc += sc[s] * bf2f(v[(size_t)(b * S_ + s) * D_ + h * DH_ + d]);
  red[tid] = acc; __syncthreads();
  if (tid < DH_) {
    float o = (red[tid] + red[tid + 64] + red[tid + 128] + red[tid + 192]) * inv;
    attn[(size_t)(b * S_ + t) * D_ + h * DH_ + tid] = f2bf(o);
  }
}

// ---- candidate tokens: 3-key softmax; one wave per (b,k,h) ----
__global__ __launch_bounds__(256)
void attn_cand(const unsigned short* __restrict__ q, const unsigned short* __restrict__ k,
               const unsigned short* __restrict__ v, const int* __restrict__ cmask,
               unsigned short* __restrict__ attn)
{
  int gid  = blockIdx.x * 8 + (threadIdx.x >> 5);   // b*(K*H) + kk*H + h
  int lane = threadIdx.x & 31;
  int h  = gid & (H_ - 1);
  int kk = (gid >> 2) & (K_ - 1);
  int b  = gid >> 12;                                // / (K*H)
  size_t rc = (size_t)(b * S_ + 2 + kk) * D_ + h * DH_;
  size_t r0 = (size_t)(b * S_    ) * D_ + h * DH_;
  size_t r1 = (size_t)(b * S_ + 1) * D_ + h * DH_;

  float q0 = bf2f(q[rc + lane]), q1 = bf2f(q[rc + lane + 32]);
  float p0 = q0 * bf2f(k[r0 + lane]) + q1 * bf2f(k[r0 + lane + 32]);
  float p1 = q0 * bf2f(k[r1 + lane]) + q1 * bf2f(k[r1 + lane + 32]);
  float p2 = q0 * bf2f(k[rc + lane]) + q1 * bf2f(k[rc + lane + 32]);
  #pragma unroll
  for (int off = 16; off >= 1; off >>= 1) {
    p0 += __shfl_xor(p0, off, 32);
    p1 += __shfl_xor(p1, off, 32);
    p2 += __shfl_xor(p2, off, 32);
  }
  p0 *= SCALE_; p1 *= SCALE_; p2 *= SCALE_;
  if (cmask[b * K_ + kk] == 0) p2 = -1e30f;
  float m  = fmaxf(p0, fmaxf(p1, p2));
  float e0 = __expf(p0 - m), e1 = __expf(p1 - m), e2 = __expf(p2 - m);
  float inv = 1.f / (e0 + e1 + e2);
  float w0 = e0 * inv, w1 = e1 * inv, w2 = e2 * inv;
  float o0 = w0 * bf2f(v[r0 + lane])      + w1 * bf2f(v[r1 + lane])      + w2 * bf2f(v[rc + lane]);
  float o1 = w0 * bf2f(v[r0 + lane + 32]) + w1 * bf2f(v[r1 + lane + 32]) + w2 * bf2f(v[rc + lane + 32]);
  attn[rc + lane]      = f2bf(o0);
  attn[rc + lane + 32] = f2bf(o1);
}

// ---- head: fused LN + dot(head_w) + mask; one wave per output element ----
__global__ __launch_bounds__(256)
void head_kernel(const float* __restrict__ x, const float* __restrict__ w,
                 const float* __restrict__ bb, const float* __restrict__ hw,
                 const float* __restrict__ hb, const int* __restrict__ cmask,
                 float* __restrict__ out)
{
  int wid  = (blockIdx.x * 256 + threadIdx.x) >> 5;
  int lane = threadIdx.x & 31;
  const int rows = B_ * (K_ + 1);
  if (wid >= rows) return;
  int b = wid / (K_ + 1);
  int j = wid % (K_ + 1);
  int s = (j == 0) ? 1 : (j + 1);                // out[:,0]=dustbin token logit
  const float* xr = x + (size_t)(b * S_ + s) * D_;
  float vals[8], sm = 0.f, sq = 0.f;
  #pragma unroll
  for (int i = 0; i < 8; ++i) { float t = xr[lane + 32 * i]; vals[i] = t; sm += t; sq += t * t; }
  #pragma unroll
  for (int off = 16; off >= 1; off >>= 1) { sm += __shfl_xor(sm, off, 32); sq += __shfl_xor(sq, off, 32); }
  float mean = sm * (1.f / D_);
  float var  = sq * (1.f / D_) - mean * mean;
  float inv  = rsqrtf(var + 1e-5f);
  float dot = 0.f;
  #pragma unroll
  for (int i = 0; i < 8; ++i) {
    int d = lane + 32 * i;
    dot += ((vals[i] - mean) * inv * w[d] + bb[d]) * hw[d];
  }
  #pragma unroll
  for (int off = 16; off >= 1; off >>= 1) dot += __shfl_xor(dot, off, 32);
  if (lane == 0) {
    float logit = dot + hb[0];
    if (j > 0 && cmask[b * K_ + (j - 1)] == 0) logit = -INFINITY;
    out[wid] = logit;
  }
}

// -------------------------------------------------------------------------
extern "C" void kernel_launch(void* const* d_in, const int* in_sizes, int n_in,
                              void* d_out, int out_size, void* d_ws, size_t ws_size,
                              hipStream_t stream)
{
  (void)in_sizes; (void)n_in; (void)out_size; (void)ws_size;
  const float* query      = (const float*)d_in[0];
  const float* candidates = (const float*)d_in[1];
  const int*   cmask      = (const int*)  d_in[2];
  const float* qp_w   = (const float*)d_in[3];
  const float* qp_b   = (const float*)d_in[4];
  const float* cp_w   = (const float*)d_in[5];
  const float* cp_b   = (const float*)d_in[6];
  const float* dustbin= (const float*)d_in[7];
  const float* te     = (const float*)d_in[8];
  const float* ln1_w  = (const float*)d_in[9];
  const float* ln1_b  = (const float*)d_in[10];
  const float* ln2_w  = (const float*)d_in[11];
  const float* ln2_b  = (const float*)d_in[12];
  const float* Wq     = (const float*)d_in[13];
  const float* bq     = (const float*)d_in[14];
  const float* Wk     = (const float*)d_in[15];
  const float* bk     = (const float*)d_in[16];
  const float* Wv     = (const float*)d_in[17];
  const float* bv     = (const float*)d_in[18];
  const float* Wo     = (const float*)d_in[19];
  const float* bo     = (const float*)d_in[20];
  const float* ff1_w  = (const float*)d_in[21];
  const float* ff1_b  = (const float*)d_in[22];
  const float* ff2_w  = (const float*)d_in[23];
  const float* ff2_b  = (const float*)d_in[24];
  const float* hln_w  = (const float*)d_in[25];
  const float* hln_b  = (const float*)d_in[26];
  const float* head_w = (const float*)d_in[27];
  const float* head_b = (const float*)d_in[28];

  // workspace layout
  char* ws = (char*)d_ws;
  size_t off = 0;
  float* x = (float*)(ws + off);                 off += (size_t)MS_ * D_ * 4;   // residual stream (fp32)
  unsigned short* lnb   = (unsigned short*)(ws + off); off += (size_t)MS_ * D_ * 2;
  unsigned short* qbuf  = (unsigned short*)(ws + off); off += (size_t)MS_ * D_ * 2;
  unsigned short* kbuf  = (unsigned short*)(ws + off); off += (size_t)MS_ * D_ * 2;
  unsigned short* vbuf  = (unsigned short*)(ws + off); off += (size_t)MS_ * D_ * 2;
  unsigned short* attnb = (unsigned short*)(ws + off); off += (size_t)MS_ * D_ * 2;
  unsigned short* hbuf  = (unsigned short*)(ws + off); off += (size_t)MS_ * FF_ * 2;

  // 1) token embed: query/dustbin rows (tiny) + candidate rows (big WMMA GEMM,
  //    rows remapped (b*K + k) -> (b*S + 2 + k), epilogue adds cp_b + type_emb[2])
  embed_small<<<B_, 256, 0, stream>>>(query, qp_w, qp_b, dustbin, te, x);
  gemm_wmma<false, 1, false, true><<<dim3(D_ / 64, (B_ * K_) / 128), 256, 0, stream>>>(
      candidates, cp_w, cp_b, te + 2 * D_, nullptr, x, D_, C_);

  for (int l = 0; l < L_; ++l) {
    // LN1 -> bf16
    ln_rows<<<MS_ / 8, 256, 0, stream>>>(x, ln1_w + l * D_, ln1_b + l * D_, lnb, MS_);
    // QKV projections (bf16 out)
    gemm_wmma<true, 0, true, false><<<dim3(D_ / 64, MS_ / 128), 256, 0, stream>>>(
        lnb, Wq + (size_t)l * D_ * D_, bq + l * D_, nullptr, nullptr, qbuf, D_, D_);
    gemm_wmma<true, 0, true, false><<<dim3(D_ / 64, MS_ / 128), 256, 0, stream>>>(
        lnb, Wk + (size_t)l * D_ * D_, bk + l * D_, nullptr, nullptr, kbuf, D_, D_);
    gemm_wmma<true, 0, true, false><<<dim3(D_ / 64, MS_ / 128), 256, 0, stream>>>(
        lnb, Wv + (size_t)l * D_ * D_, bv + l * D_, nullptr, nullptr, vbuf, D_, D_);
    // star attention
    attn_global<<<dim3(2, H_, B_), 256, 0, stream>>>(qbuf, kbuf, vbuf, cmask, attnb);
    attn_cand<<<(B_ * K_ * H_) / 8, 256, 0, stream>>>(qbuf, kbuf, vbuf, cmask, attnb);
    // output projection + residual into x
    gemm_wmma<true, 3, false, false><<<dim3(D_ / 64, MS_ / 128), 256, 0, stream>>>(
        attnb, Wo + (size_t)l * D_ * D_, bo + l * D_, nullptr, x, x, D_, D_);
    // LN2 -> bf16, FFN: gelu(h@ff1) @ ff2 + residual
    ln_rows<<<MS_ / 8, 256, 0, stream>>>(x, ln2_w + l * D_, ln2_b + l * D_, lnb, MS_);
    gemm_wmma<true, 2, true, false><<<dim3(FF_ / 64, MS_ / 128), 256, 0, stream>>>(
        lnb, ff1_w + (size_t)l * FF_ * D_, ff1_b + l * FF_, nullptr, nullptr, hbuf, FF_, D_);
    gemm_wmma<true, 3, false, false><<<dim3(D_ / 64, MS_ / 128), 256, 0, stream>>>(
        hbuf, ff2_w + (size_t)l * D_ * FF_, ff2_b + l * D_, nullptr, x, x, D_, FF_);
  }

  // head: LN + dot + mask -> d_out (B, K+1) fp32
  head_kernel<<<(B_ * (K_ + 1)) / 8, 256, 0, stream>>>(
      x, hln_w, hln_b, head_w, head_b, cmask, (float*)d_out);
}
